// TridentDecoder_3590592659519
// MI455X (gfx1250) — matrix operand order repaired and checked by reading, not compiled, not run on hardware
//
#include <hip/hip_runtime.h>

// ---------------------------------------------------------------------------
// TridentDecoder for MI455X (gfx1250, wave32, WMMA).
// Mixed precision: bf16 operands, f32 accumulation via v_wmma_f32_16x16x32_bf16.
// GEMM is double-buffered in LDS and software-pipelined: next K-tile's global
// loads are issued before the current tile's WMMAs, so s_wait_loadcnt lands
// after the matrix math. All operand rows are 16B-aligned (W2v padded to
// 32128 columns) -> staging is pure b128 traffic.
// ---------------------------------------------------------------------------

typedef __attribute__((ext_vector_type(16))) __bf16       bf16x16;
typedef __attribute__((ext_vector_type(8)))  float        f32x8;
typedef __attribute__((ext_vector_type(4)))  unsigned int u32x4;

#define TD_DEPTH 11
#define TD_H     1024
#define TD_V     32001
#define TD_VP    32128     // V padded to a multiple of 128 (16B-aligned bf16 rows)
#define TD_NODES 4095      // 2^(DEPTH+1) - 1

#define GF_SIGMOID  1      // apply sigmoid in epilogue
#define GF_OUT_BF16 2      // store bf16 to Cb (else f32 to Cf)
#define GF_VOCAB    4      // store f32 to Cf at permuted row (rowmap)

// ---------------------------------------------------------------------------
// f32 -> bf16 grid-stride converter (flat)
// ---------------------------------------------------------------------------
__global__ void td_f32_to_bf16(const float* __restrict__ s,
                               __bf16* __restrict__ d, long n) {
  long i      = (long)blockIdx.x * blockDim.x + threadIdx.x;
  long stride = (long)gridDim.x * blockDim.x;
  for (; i < n; i += stride) d[i] = (__bf16)s[i];
}

// ---------------------------------------------------------------------------
// f32 -> bf16 pitched converter with zero-filled column padding.
// Repacks W2v (2048 x 32001) into a 2048 x 32128 bf16 matrix.
// ---------------------------------------------------------------------------
__global__ void td_f32_to_bf16_pitched(const float* __restrict__ s, int lds_,
                                       __bf16* __restrict__ d, int ldd,
                                       int rows, int cols) {
  long total  = (long)rows * ldd;
  long i      = (long)blockIdx.x * blockDim.x + threadIdx.x;
  long stride = (long)gridDim.x * blockDim.x;
  for (; i < total; i += stride) {
    int r = (int)(i / ldd);
    int c = (int)(i - (long)r * ldd);
    d[i] = (c < cols) ? (__bf16)s[(size_t)r * lds_ + c] : (__bf16)0.0f;
  }
}

// ---------------------------------------------------------------------------
// BFS row -> pre-order row.  Node i (BFS, 0-based): depth d = log2(i+1),
// in-level index p.  Each right turn skips a left subtree of 2^(DEPTH-k)-1.
// ---------------------------------------------------------------------------
__global__ void td_build_rowmap(int* __restrict__ rm) {
  int i = blockIdx.x * blockDim.x + threadIdx.x;
  if (i >= TD_NODES) return;
  int n = i + 1;
  int d = 31 - __clz(n);
  int p = n - (1 << d);
  int pos = 0;
  for (int k = 0; k < d; ++k) {
    int bit = (p >> (d - 1 - k)) & 1;
    pos += 1 + bit * ((1 << (TD_DEPTH - k)) - 1);
  }
  rm[i] = pos;
}

// ---------------------------------------------------------------------------
// Generic tiled GEMM:  C = act(A[MxK] * B[KxN] + bias)
//   block = 256 threads = 8 waves; tile 128x128, K-step 32, LDS double-buffer.
//   wave (wm,wn): wm in {0,1} -> 64 rows, wn in {0..3} -> 32 cols,
//   i.e. 4x2 = 8 x v_wmma_f32_16x16x32_bf16 per wave per K-step.
// Requirements: K % 32 == 0; ldb % 8 == 0 and B allocated with
// gridDim.y*128 readable columns per row (pad columns are zero).
// A rows are clamped (not guarded): OOB C rows are dropped at the store.
// ---------------------------------------------------------------------------
__global__ void __launch_bounds__(256)
td_gemm_bf16(const __bf16* __restrict__ A, int lda,
             const __bf16* __restrict__ B, int ldb,
             const float*  __restrict__ bias,
             __bf16* __restrict__ Cb, float* __restrict__ Cf, int ldc,
             const int* __restrict__ rowmap,
             int M, int N, int K, int flags) {
  __shared__ __bf16 As[2 * 128 * 32];  // [buf][m][k]
  __shared__ __bf16 Bs[2 * 32 * 128];  // [buf][k][n]

  const int tid  = threadIdx.x;
  const int lane = tid & 31;
  const int wid  = tid >> 5;
  const int wm   = wid & 1;
  const int wn   = wid >> 1;
  const int lr   = lane & 15;
  const int g    = lane >> 4;

  const int tileM = blockIdx.x * 128;
  const int tileN = blockIdx.y * 128;

  f32x8 acc[4][2];
#pragma unroll
  for (int s = 0; s < 4; ++s)
#pragma unroll
    for (int t = 0; t < 2; ++t)
#pragma unroll
      for (int r = 0; r < 8; ++r) acc[s][t][r] = 0.0f;

  const int am  = tid >> 1;        // 0..127  (A row in tile)
  const int ak0 = (tid & 1) << 4;  // 0 | 16  (A k-chunk)
  const int bk  = tid >> 3;        // 0..31   (B k row)
  const int bn0 = (tid & 7) << 4;  // 0..112  (B n chunk)

  // Clamp the A row (branch-free staging); clamped rows produce duplicate
  // partial sums that are discarded by the row<M store guard.
  int arow = tileM + am;
  if (arow > M - 1) arow = M - 1;
  const __bf16* aptr = A + (size_t)arow * lda + ak0;
  const __bf16* bptr = B + (size_t)bk * ldb + tileN + bn0;

  // ---- preload K-tile 0 ----
  u32x4 a0, a1, b0, b1;
  {
    const u32x4* asrc = (const u32x4*)aptr;
    a0 = asrc[0];
    a1 = asrc[1];
    const u32x4* bsrc = (const u32x4*)bptr;
    b0 = bsrc[0];
    b1 = bsrc[1];
    u32x4* adst = (u32x4*)(As + am * 32 + ak0);
    adst[0] = a0;
    adst[1] = a1;
    u32x4* bdst = (u32x4*)(Bs + bk * 128 + bn0);
    bdst[0] = b0;
    bdst[1] = b1;
  }
  __syncthreads();

  int buf = 0;
  for (int kb = 0; kb < K; kb += 32) {
    const bool have_next = (kb + 32) < K;

    // ---- issue next tile's global loads (no wait yet) ----
    if (have_next) {
      const u32x4* asrc = (const u32x4*)(aptr + kb + 32);
      a0 = asrc[0];
      a1 = asrc[1];
      const u32x4* bsrc = (const u32x4*)(bptr + (size_t)(kb + 32) * ldb);
      b0 = bsrc[0];
      b1 = bsrc[1];
      if (kb + 64 < K) {  // speculative prefetch two tiles ahead
        __builtin_prefetch(aptr + kb + 64, 0, 1);
        __builtin_prefetch(bptr + (size_t)(kb + 64) * ldb, 0, 1);
      }
    }

    // ---- compute on current buffer (ds_load + wmma; hides the loads) ----
    const __bf16* Asb = As + buf * (128 * 32);
    const __bf16* Bsb = Bs + buf * (32 * 128);
    bf16x16 afrag[4], bfrag[2];
#pragma unroll
    for (int t = 0; t < 2; ++t) {
      // B: lane holds N col = lr; K rows [8g,8g+8) and [16+8g,16+8g+8)
      const __bf16* p = Bsb + wn * 32 + t * 16 + lr;
      bf16x16 v;
#pragma unroll
      for (int e = 0; e < 16; ++e)
        v[e] = p[((e >> 3) * 16 + g * 8 + (e & 7)) * 128];
      bfrag[t] = v;
    }
#pragma unroll
    for (int s = 0; s < 4; ++s) {
      // A: lane holds row M=lr; K half-words [8g,8g+8) and [16+8g,16+8g+8)
      const u32x4* p = (const u32x4*)(Asb + (wm * 64 + s * 16 + lr) * 32);
      union { u32x4 q[2]; bf16x16 v; } u;
      u.q[0] = p[g];
      u.q[1] = p[2 + g];
      afrag[s] = u.v;
    }
#pragma unroll
    for (int s = 0; s < 4; ++s)
#pragma unroll
      for (int t = 0; t < 2; ++t)
        acc[s][t] = __builtin_amdgcn_wmma_f32_16x16x32_bf16(
            false, afrag[s], false, bfrag[t], (short)0, acc[s][t], false, false);

    // ---- drain next tile's loads into the other buffer ----
    if (have_next) {
      __bf16* Asn = As + (buf ^ 1) * (128 * 32);
      __bf16* Bsn = Bs + (buf ^ 1) * (32 * 128);
      u32x4* adst = (u32x4*)(Asn + am * 32 + ak0);
      adst[0] = a0;
      adst[1] = a1;
      u32x4* bdst = (u32x4*)(Bsn + bk * 128 + bn0);
      bdst[0] = b0;
      bdst[1] = b1;
    }
    __syncthreads();
    buf ^= 1;
  }

  // ---- epilogue: C/D layout M = r + 8*g, N = lr ----
#pragma unroll
  for (int s = 0; s < 4; ++s) {
#pragma unroll
    for (int r = 0; r < 8; ++r) {
      const int row = tileM + wm * 64 + s * 16 + g * 8 + r;
      if (row >= M) continue;
      const size_t outrow =
          (flags & GF_VOCAB) ? (size_t)rowmap[row] : (size_t)row;
#pragma unroll
      for (int t = 0; t < 2; ++t) {
        const int col = tileN + wn * 32 + t * 16 + lr;
        if (col < N) {
          float v = acc[s][t][r];
          if (bias) v += bias[col];
          if (flags & GF_SIGMOID) v = 1.0f / (1.0f + __expf(-v));
          if (flags & GF_VOCAB) {
            Cf[outrow * N + col] = v;  // pre-order permuted store
          } else if (flags & GF_OUT_BF16) {
            Cb[outrow * ldc + col] = (__bf16)v;
          } else {
            Cf[outrow * ldc + col] = v;
          }
        }
      }
    }
  }
}

// ---------------------------------------------------------------------------
// In-place row log-softmax. One block per row; the whole 128 KB row is staged
// in dynamic LDS (WGP has 320 KB) so d_out is read once and written once.
// ---------------------------------------------------------------------------
__global__ void __launch_bounds__(256)
td_logsoftmax(float* __restrict__ out) {
  extern __shared__ float srow[];
  __shared__ float red[8];
  float* p = out + (size_t)blockIdx.x * TD_V;
  const int tid = threadIdx.x;

  float lmax = -INFINITY;
  for (int i = tid; i < TD_V; i += 256) {
    float x = p[i];
    srow[i] = x;
    lmax = fmaxf(lmax, x);
  }
#pragma unroll
  for (int m = 16; m > 0; m >>= 1) lmax = fmaxf(lmax, __shfl_xor(lmax, m, 32));
  if ((tid & 31) == 0) red[tid >> 5] = lmax;
  __syncthreads();
  float vmax = red[0];
#pragma unroll
  for (int w = 1; w < 8; ++w) vmax = fmaxf(vmax, red[w]);
  __syncthreads();

  float lsum = 0.0f;
  for (int i = tid; i < TD_V; i += 256) lsum += __expf(srow[i] - vmax);
#pragma unroll
  for (int m = 16; m > 0; m >>= 1) lsum += __shfl_xor(lsum, m, 32);
  if ((tid & 31) == 0) red[tid >> 5] = lsum;
  __syncthreads();
  float s = 0.0f;
#pragma unroll
  for (int w = 0; w < 8; ++w) s += red[w];
  const float lse = vmax + __logf(s);

  for (int i = tid; i < TD_V; i += 256) p[i] = srow[i] - lse;
}

// ---------------------------------------------------------------------------
extern "C" void kernel_launch(void* const* d_in, const int* in_sizes, int n_in,
                              void* d_out, int out_size, void* d_ws, size_t ws_size,
                              hipStream_t stream) {
  (void)in_sizes; (void)n_in; (void)out_size; (void)ws_size;

  const float* root = (const float*)d_in[0];
  const float* W1v  = (const float*)d_in[1];
  const float* b1v  = (const float*)d_in[2];
  const float* W2v  = (const float*)d_in[3];
  const float* b2v  = (const float*)d_in[4];
  const float* W1c  = (const float*)d_in[5];
  const float* b1c  = (const float*)d_in[6];
  const float* W2c  = (const float*)d_in[7];
  const float* b2c  = (const float*)d_in[8];
  float* out = (float*)d_out;

  // ---- workspace carve-up (~174 MB) ----
  char* ws = (char*)d_ws;
  size_t off = 0;
  auto wsalloc = [&](size_t bytes) -> void* {
    void* p = (void*)(ws + off);
    off += (bytes + 255) & ~(size_t)255;
    return p;
  };
  __bf16* W1v_b  = (__bf16*)wsalloc(sizeof(__bf16) * (size_t)TD_H * 2 * TD_H);
  __bf16* W2v_b  = (__bf16*)wsalloc(sizeof(__bf16) * (size_t)2 * TD_H * TD_VP);
  __bf16* W1c_b  = (__bf16*)wsalloc(sizeof(__bf16) * (size_t)TD_H * 2 * TD_H);
  __bf16* W2c_b  = (__bf16*)wsalloc(sizeof(__bf16) * (size_t)2 * TD_H * 2 * TD_H);
  __bf16* hidden = (__bf16*)wsalloc(sizeof(__bf16) * (size_t)4096 * TD_H);      // BFS rows
  __bf16* act    = (__bf16*)wsalloc(sizeof(__bf16) * (size_t)4096 * 2 * TD_H);  // hidden1
  int*    rowmap = (int*)wsalloc(sizeof(int) * 4096);

  auto cvt = [&](const float* s, __bf16* d, long n) {
    int grid = (int)((n + 255) / 256);
    if (grid > 2048) grid = 2048;
    td_f32_to_bf16<<<grid, 256, 0, stream>>>(s, d, n);
  };
  cvt(W1v, W1v_b, (long)TD_H * 2 * TD_H);
  cvt(W1c, W1c_b, (long)TD_H * 2 * TD_H);
  cvt(W2c, W2c_b, (long)2 * TD_H * 2 * TD_H);
  cvt(root, hidden, TD_H);  // BFS row 0
  // W2v: repack with zero-padded rows so GEMM B staging can use b128 loads.
  td_f32_to_bf16_pitched<<<2048, 256, 0, stream>>>(W2v, TD_V, W2v_b, TD_VP,
                                                   2 * TD_H, TD_V);
  td_build_rowmap<<<(TD_NODES + 255) / 256, 256, 0, stream>>>(rowmap);

  auto gemm = [&](const __bf16* A, int lda, const __bf16* B, int ldb,
                  const float* bias, __bf16* Cb, float* Cf, int ldc,
                  const int* rmap, int M, int N, int K, int flags) {
    dim3 grid((M + 127) / 128, (N + 127) / 128);
    td_gemm_bf16<<<grid, 256, 0, stream>>>(A, lda, B, ldb, bias, Cb, Cf, ldc,
                                           rmap, M, N, K, flags);
  };

  // ---- tree expansion: level d (2^d rows) -> level d+1 (2^(d+1) rows) ----
  // (M, 2H) child output is bitwise the (2M, H) child-hidden block.
  for (int d = 0; d < TD_DEPTH; ++d) {
    const int Mrows = 1 << d;
    const __bf16* hsrc = hidden + (size_t)((1 << d) - 1) * TD_H;
    __bf16* hdst       = hidden + (size_t)((1 << (d + 1)) - 1) * TD_H;
    gemm(hsrc, TD_H, W1c_b, 2 * TD_H, b1c, act, nullptr, 2 * TD_H, nullptr,
         Mrows, 2 * TD_H, TD_H, GF_SIGMOID | GF_OUT_BF16);
    gemm(act, 2 * TD_H, W2c_b, 2 * TD_H, b2c, hdst, nullptr, 2 * TD_H, nullptr,
         Mrows, 2 * TD_H, 2 * TD_H, GF_OUT_BF16);
  }

  // ---- vocab head over all 4095 nodes (the 537-GFLOP part) ----
  gemm(hidden, TD_H, W1v_b, 2 * TD_H, b1v, act, nullptr, 2 * TD_H, nullptr,
       TD_NODES, 2 * TD_H, TD_H, GF_SIGMOID | GF_OUT_BF16);
  gemm(act, 2 * TD_H, W2v_b, TD_VP, b2v, nullptr, out, TD_V, rowmap,
       TD_NODES, TD_V, 2 * TD_H, GF_VOCAB);

  // ---- in-place log-softmax, full row staged in LDS ----
  td_logsoftmax<<<TD_NODES, 256, TD_V * sizeof(float), stream>>>(out);
}
